// EdgeFeatureNetwork_20229295964755
// MI455X (gfx1250) — matrix operand order repaired and checked by reading, not compiled, not run on hardware
//
#include <hip/hip_runtime.h>

// ---- tile / layout constants -------------------------------------------------
#define DN    64      // node feature dim
#define W1KP  168     // w1t row stride in halfs (pad 160 -> 168: conflict-free)
#define W2KP  72      // w2t row stride (pad 64 -> 72)
#define W3KP  40      // w3t row stride (pad 32 -> 40)
#define H1P   72      // h1 stage row stride in halfs (pad 64 -> 72)
#define H2P   40      // h2 stage row stride in halfs (pad 32 -> 40)
#define OSP   20      // out stage row stride in floats (pad 16 -> 20)

typedef _Float16 h16 __attribute__((ext_vector_type(16)));
typedef _Float16 h8  __attribute__((ext_vector_type(8)));
typedef float    f8  __attribute__((ext_vector_type(8)));
typedef float    f4  __attribute__((ext_vector_type(4)));

static __device__ __forceinline__ f8 wmma_f16(h16 a, h16 b, f8 c) {
  // D = A(16x32 f16) * B(32x16 f16) + C(16x16 f32)
  return __builtin_amdgcn_wmma_f32_16x16x32_f16(false, a, false, b,
                                                (short)0, c, false, false);
}

// A-fragment (16x32 f16) built from a global f32 row.
// Lane<16 holds K {kb..kb+7, kb+16..kb+23}; lane>=16 holds K {kb+8..kb+15, kb+24..kb+31}.
static __device__ __forceinline__ h16 load_a_global(const float* __restrict__ row,
                                                    int kbase, int half) {
  const int g0 = kbase + half * 8;
  const int g1 = g0 + 16;
  f4 a0 = *(const f4*)(row + g0);
  f4 a1 = *(const f4*)(row + g0 + 4);
  f4 a2 = *(const f4*)(row + g1);
  f4 a3 = *(const f4*)(row + g1 + 4);
  h16 r;
#pragma unroll
  for (int i = 0; i < 4; ++i) {
    r[i]      = (_Float16)a0[i];
    r[4 + i]  = (_Float16)a1[i];
    r[8 + i]  = (_Float16)a2[i];
    r[12 + i] = (_Float16)a3[i];
  }
  return r;
}

// A-fragment from an LDS f16 row (same per-lane K split as above).
static __device__ __forceinline__ h16 load_a_lds(const _Float16* row,
                                                 int kbase, int half) {
  const _Float16* p0 = row + kbase + half * 8;
  h8 a0 = *(const h8*)(p0);
  h8 a1 = *(const h8*)(p0 + 16);
  h16 r;
#pragma unroll
  for (int i = 0; i < 8; ++i) { r[i] = a0[i]; r[8 + i] = a1[i]; }
  return r;
}

// B-fragment (32x16 f16) from an LDS f16 row of W^T[n][k].
// Lane<16 holds K {kb..kb+7, kb+16..kb+23}; lane>=16 holds K {kb+8..kb+15, kb+24..kb+31}.
static __device__ __forceinline__ h16 load_b_lds(const _Float16* row,
                                                 int kbase, int half) {
  const _Float16* p = row + kbase + half * 8;
  h8 b0 = *(const h8*)(p);
  h8 b1 = *(const h8*)(p + 16);
  h16 r;
#pragma unroll
  for (int i = 0; i < 8; ++i) { r[i] = b0[i]; r[8 + i] = b1[i]; }
  return r;
}

__global__ __launch_bounds__(256)
void edge_mlp_wmma_kernel(const float* __restrict__ nf,
                          const long long* __restrict__ ei,
                          const float* __restrict__ spat,
                          const float* __restrict__ W1, const float* __restrict__ b1,
                          const float* __restrict__ W2, const float* __restrict__ b2,
                          const float* __restrict__ W3, const float* __restrict__ b3,
                          float* __restrict__ out,
                          int E, int numTiles) {
  __shared__ _Float16 w1t[64 * W1KP];                 // W1^T [n][k], k padded 130->160(168)
  __shared__ _Float16 w2t[32 * W2KP];                 // W2^T [n][k]
  __shared__ _Float16 w3t[16 * W3KP];                 // W3^T [n][k]
  __shared__ __align__(16) _Float16 stageAll[8][16 * H1P];  // per-wave activation stage

  const int tid = threadIdx.x;

  // ---- stage weights into LDS: zero-fill (K padding), then transposed f16 copy
  for (int i = tid; i < 64 * W1KP; i += 256) w1t[i] = (_Float16)0.0f;
  for (int i = tid; i < 32 * W2KP; i += 256) w2t[i] = (_Float16)0.0f;
  for (int i = tid; i < 16 * W3KP; i += 256) w3t[i] = (_Float16)0.0f;
  __syncthreads();
  for (int i = tid; i < 130 * 64; i += 256) {         // W1 is [130][64] row-major
    int k = i >> 6, n = i & 63;
    w1t[n * W1KP + k] = (_Float16)W1[i];
  }
  for (int i = tid; i < 64 * 32; i += 256) {          // W2 [64][32]
    int k = i >> 5, n = i & 31;
    w2t[n * W2KP + k] = (_Float16)W2[i];
  }
  for (int i = tid; i < 32 * 16; i += 256) {          // W3 [32][16]
    int k = i >> 4, n = i & 15;
    w3t[n * W3KP + k] = (_Float16)W3[i];
  }
  __syncthreads();

  const int lane = tid & 31;
  const int wave = tid >> 5;
  const int m    = lane & 15;    // row within tile / output column
  const int half = lane >> 4;
  _Float16* stage = stageAll[wave];
  float*    fstage = (float*)stage;

  // per-lane bias values (bias depends only on output column n = lane%16)
  const float b1v0 = b1[m], b1v1 = b1[16 + m], b1v2 = b1[32 + m], b1v3 = b1[48 + m];
  const float b2v0 = b2[m], b2v1 = b2[16 + m];
  const float b3v  = b3[m];

  const int waveStride = gridDim.x * 8;
  for (int tile = blockIdx.x * 8 + wave; tile < numTiles; tile += waveStride) {
    const int eBase = tile * 16;
    const int e  = eBase + m;
    const int eC = (e < E) ? e : (E - 1);

    const long long sidx = ei[eC];
    const long long didx = ei[(long long)E + eC];
    const float* srow = nf + sidx * DN;
    const float* drow = nf + didx * DN;

    // ---------------- layer 1: [16 x 130] @ [130 x 64] ----------------
    f8 acc0, acc1, acc2, acc3;
#pragma unroll
    for (int r = 0; r < 8; ++r) { acc0[r] = b1v0; acc1[r] = b1v1; acc2[r] = b1v2; acc3[r] = b1v3; }

#pragma unroll
    for (int s = 0; s < 4; ++s) {
      const float* row = (s < 2) ? srow : drow;       // src rows then dst rows
      const int kb  = (s & 1) * 32;                   // column block within node row
      const int wk  = s * 32;                         // K block within W1
      h16 a = load_a_global(row, kb, half);
      h16 bf;
      bf = load_b_lds(w1t + (0 * 16 + m) * W1KP, wk, half); acc0 = wmma_f16(a, bf, acc0);
      bf = load_b_lds(w1t + (1 * 16 + m) * W1KP, wk, half); acc1 = wmma_f16(a, bf, acc1);
      bf = load_b_lds(w1t + (2 * 16 + m) * W1KP, wk, half); acc2 = wmma_f16(a, bf, acc2);
      bf = load_b_lds(w1t + (3 * 16 + m) * W1KP, wk, half); acc3 = wmma_f16(a, bf, acc3);
    }
    { // spatial K-block: K 128..129 live, rest zero-padded (w1t rows 130.. are zero)
      h16 a;
#pragma unroll
      for (int i = 0; i < 16; ++i) a[i] = (_Float16)0.0f;
      if (half == 0) {
        a[0] = (_Float16)spat[(size_t)eC * 2 + 0];
        a[1] = (_Float16)spat[(size_t)eC * 2 + 1];
      }
      h16 bf;
      bf = load_b_lds(w1t + (0 * 16 + m) * W1KP, 128, half); acc0 = wmma_f16(a, bf, acc0);
      bf = load_b_lds(w1t + (1 * 16 + m) * W1KP, 128, half); acc1 = wmma_f16(a, bf, acc1);
      bf = load_b_lds(w1t + (2 * 16 + m) * W1KP, 128, half); acc2 = wmma_f16(a, bf, acc2);
      bf = load_b_lds(w1t + (3 * 16 + m) * W1KP, 128, half); acc3 = wmma_f16(a, bf, acc3);
    }

    // ReLU + stage h1 as f16 [16 rows][H1P] (D-frag col-per-lane -> row-major LDS)
#pragma unroll
    for (int r = 0; r < 8; ++r) {
      const int M = r + half * 8;
      stage[M * H1P +      m] = (_Float16)fmaxf(acc0[r], 0.0f);
      stage[M * H1P + 16 + m] = (_Float16)fmaxf(acc1[r], 0.0f);
      stage[M * H1P + 32 + m] = (_Float16)fmaxf(acc2[r], 0.0f);
      stage[M * H1P + 48 + m] = (_Float16)fmaxf(acc3[r], 0.0f);
    }

    // ---------------- layer 2: [16 x 64] @ [64 x 32] ----------------
    f8 g0, g1;
#pragma unroll
    for (int r = 0; r < 8; ++r) { g0[r] = b2v0; g1[r] = b2v1; }
#pragma unroll
    for (int s = 0; s < 2; ++s) {
      const int kb2 = s * 32;
      h16 a = load_a_lds(stage + m * H1P, kb2, half);
      h16 bf;
      bf = load_b_lds(w2t +        m * W2KP, kb2, half); g0 = wmma_f16(a, bf, g0);
      bf = load_b_lds(w2t + (16 + m) * W2KP, kb2, half); g1 = wmma_f16(a, bf, g1);
    }
#pragma unroll
    for (int r = 0; r < 8; ++r) {
      const int M = r + half * 8;
      stage[M * H2P +      m] = (_Float16)fmaxf(g0[r], 0.0f);
      stage[M * H2P + 16 + m] = (_Float16)fmaxf(g1[r], 0.0f);
    }

    // ---------------- layer 3: [16 x 32] @ [32 x 16] ----------------
    f8 o;
#pragma unroll
    for (int r = 0; r < 8; ++r) o[r] = b3v;
    {
      h16 a  = load_a_lds(stage + m * H2P, 0, half);
      h16 bf = load_b_lds(w3t + m * W3KP, 0, half);
      o = wmma_f16(a, bf, o);
    }

    // transpose result through LDS -> coalesced 2x b128 stores per lane
#pragma unroll
    for (int r = 0; r < 8; ++r)
      fstage[(r + half * 8) * OSP + m] = o[r];

    f4 o0 = *(const f4*)(fstage + m * OSP + half * 8);
    f4 o1 = *(const f4*)(fstage + m * OSP + half * 8 + 4);
    if (e < E) {
      float* op = out + (size_t)e * 16 + half * 8;
      *(f4*)op       = o0;
      *((f4*)op + 1) = o1;
    }
  }
}

extern "C" void kernel_launch(void* const* d_in, const int* in_sizes, int n_in,
                              void* d_out, int out_size, void* d_ws, size_t ws_size,
                              hipStream_t stream) {
  const float*     nf   = (const float*)d_in[0];
  const long long* ei   = (const long long*)d_in[1];   // int64 edge_index [2][E]
  const float*     spat = (const float*)d_in[2];
  const float*     W1   = (const float*)d_in[3];
  const float*     b1   = (const float*)d_in[4];
  const float*     W2   = (const float*)d_in[5];
  const float*     b2   = (const float*)d_in[6];
  const float*     W3   = (const float*)d_in[7];
  const float*     b3   = (const float*)d_in[8];
  float*           out  = (float*)d_out;

  const int E        = in_sizes[1] / 2;
  const int numTiles = (E + 15) / 16;

  int grid = 512;                       // 4096 waves, grid-stride over 50k tiles
  if (grid * 8 > numTiles) grid = (numTiles + 7) / 8;

  edge_mlp_wmma_kernel<<<grid, 256, 0, stream>>>(nf, ei, spat,
                                                 W1, b1, W2, b2, W3, b3,
                                                 out, E, numTiles);
}
